// Block_21792664060277
// MI455X (gfx1250) — matrix operand order repaired and checked by reading, not compiled, not run on hardware
//
#include <hip/hip_runtime.h>

// ---------------- sizes ----------------
#define BN   256
#define SEQ  197
#define CH   192
#define E3   576          // 3*CH
#define NE   1024         // edges
#define NH   3
#define HD   64
#define HID  768
#define TOK  (BN*SEQ)     // 50432, divisible by 16 -> 3152 tiles
#define NTILES_TOK (TOK/16)
#define NPAD 224          // keys padded to 7*32 for WMMA K loop
#define AGG_ELEMS (BN*SEQ*CH) // 9,682,944

// ---------------- WMMA types ----------------
typedef __bf16 bf16t;
typedef bf16t v16bf __attribute__((ext_vector_type(16)));
typedef float v8f   __attribute__((ext_vector_type(8)));

union FragU { v16bf v; uint4 q[2]; };

__device__ __forceinline__ uint4 ld16u(const unsigned short* p) {
    return *(const uint4*)p;
}

__device__ __forceinline__ unsigned short f2bf(float x) {
    unsigned int u = __float_as_uint(x);
    u += 0x7FFFu + ((u >> 16) & 1u);
    return (unsigned short)(u >> 16);
}

// A fragment: 16 x 32 bf16 tile from row-major [16][stride], rows = sub
__device__ __forceinline__ v16bf loadA(const unsigned short* base, int stride, int k0, int lane) {
    int sub = lane & 15, hi = lane >> 4;
    const unsigned short* r = base + sub * stride + k0 + hi * 8;
    FragU f;
    f.q[0] = ld16u(r);
    f.q[1] = ld16u(r + 16);
    return f.v;
}

// B fragment: 32 x 16 bf16 tile; source is B^T row-major [16 rows=n][stride, cols=k]
__device__ __forceinline__ v16bf loadB(const unsigned short* base, int stride, int k0, int lane) {
    int sub = lane & 15, hi = lane >> 4;
    const unsigned short* r = base + sub * stride + k0 + hi * 8;
    FragU f;
    f.q[0] = ld16u(r);
    f.q[1] = ld16u(r + 16);
    return f.v;
}

#define WMMA_BF16(a, b, c) \
    __builtin_amdgcn_wmma_f32_16x16x32_bf16(false, (a), false, (b), (short)0, (c), false, false)

#define V8F_ZERO {0.f,0.f,0.f,0.f,0.f,0.f,0.f,0.f}

// ---------------- TDM availability ----------------
#if defined(__gfx1250__) && __has_builtin(__builtin_amdgcn_tensor_load_to_lds) && __has_builtin(__builtin_amdgcn_s_wait_tensorcnt)
#define HAVE_TDM 1
#else
#define HAVE_TDM 0
#endif

typedef unsigned int u32x4 __attribute__((ext_vector_type(4)));
typedef int i32x4 __attribute__((ext_vector_type(4)));
typedef int i32x8 __attribute__((ext_vector_type(8)));
typedef __attribute__((address_space(3))) unsigned short lds_ushort;

// ---------------- utility kernels ----------------
__global__ void zero_kernel(float* __restrict__ p, int n) {
    int i = blockIdx.x * 256 + threadIdx.x;
    if (i < n) p[i] = 0.f;
}

__global__ void cvt_bf16_kernel(const float* __restrict__ in, unsigned short* __restrict__ out, int n) {
    int i = blockIdx.x * 256 + threadIdx.x;
    if (i < n) out[i] = f2bf(in[i]);
}

__global__ void edge_cnt_kernel(const int* __restrict__ edge, float* __restrict__ cnt) {
    int e = blockIdx.x * 256 + threadIdx.x;
    if (e < NE) atomicAdd(&cnt[edge[NE + e]], 1.0f);
}

// ---------------- LN1 + QKV GEMM ----------------
// one block = 16 tokens, 256 threads (8 waves). qkv out: bf16 [token][576]
__global__ void ln_qkv_kernel(const float* __restrict__ x,
                              const float* __restrict__ g, const float* __restrict__ b,
                              const unsigned short* __restrict__ wq,
                              unsigned short* __restrict__ qkv) {
    __shared__ unsigned short xn[16 * CH];
    __shared__ unsigned short qout[16 * E3];
    __shared__ float red[16][17];
    __shared__ float mu[16], rsg[16];

    int tid = threadIdx.x, lane = tid & 31, w = tid >> 5;
    int r = tid >> 4, c = tid & 15;
    int sub = lane & 15, hi = lane >> 4;
    size_t tok0 = (size_t)blockIdx.x * 16;
    const float* xr = x + (tok0 + r) * CH;

    float s = 0.f;
    for (int j = c; j < CH; j += 16) s += xr[j];
    red[r][c] = s; __syncthreads();
    if (tid < 16) { float m = 0.f; for (int j = 0; j < 16; j++) m += red[tid][j]; mu[tid] = m / (float)CH; }
    __syncthreads();
    float m = mu[r]; s = 0.f;
    for (int j = c; j < CH; j += 16) { float d = xr[j] - m; s += d * d; }
    red[r][c] = s; __syncthreads();
    if (tid < 16) { float v = 0.f; for (int j = 0; j < 16; j++) v += red[tid][j]; rsg[tid] = rsqrtf(v / (float)CH + 1e-5f); }
    __syncthreads();
    float inv = rsg[r];
    for (int j = c; j < CH; j += 16) xn[r * CH + j] = f2bf((xr[j] - m) * inv * g[j] + b[j]);
    __syncthreads();

    for (int t = w; t < E3 / 16; t += 8) {
        v8f acc = V8F_ZERO;
        const unsigned short* wb = wq + (size_t)(t * 16) * CH;
        #pragma unroll
        for (int kk = 0; kk < 6; kk++) {
            v16bf a = loadA(xn, CH, kk * 32, lane);
            v16bf bb = loadB(wb, CH, kk * 32, lane);
            acc = WMMA_BF16(a, bb, acc);
        }
        int nn = t * 16 + sub;
        #pragma unroll
        for (int rr = 0; rr < 8; rr++)
            qout[(hi * 8 + rr) * E3 + nn] = f2bf(acc[rr]);
    }
    __syncthreads();
    // coalesced b128 writeback
    for (int it = tid; it < 16 * (E3 / 8); it += 256) {
        int rrow = it / (E3 / 8), oct = it % (E3 / 8);
        *(uint4*)&qkv[(tok0 + rrow) * E3 + oct * 8] = *(const uint4*)&qout[rrow * E3 + oct * 8];
    }
}

// ---------------- V transpose: qkv -> vt[b][h][d][key(NPAD)] bf16 ----------------
// one block per (b,h); V tile staged to LDS by the Tensor Data Mover when available
__global__ void vt_kernel(const unsigned short* __restrict__ qkv, unsigned short* __restrict__ vt) {
    __shared__ unsigned short Vrow[SEQ * HD];   // row-major (key, d)
    int blk = blockIdx.x;
    int b = blk / NH, h = blk % NH;
    int tid = threadIdx.x;
    const unsigned short* vsrc = qkv + (size_t)b * SEQ * E3 + 2 * CH + h * HD;

#if HAVE_TDM
    if (tid < 32) {
        unsigned long long ga = (unsigned long long)vsrc;
        unsigned loff = (unsigned)(unsigned long long)(lds_ushort*)Vrow;
        u32x4 g0;
        g0[0] = 1u;                                               // count=1, no gather
        g0[1] = loff;                                             // lds_addr (bytes)
        g0[2] = (unsigned)(ga & 0xFFFFFFFFull);                   // global_addr[31:0]
        g0[3] = (unsigned)((ga >> 32) & 0x1FFFFFFull) | (2u << 30); // global_addr[56:32] | type=2
        i32x8 g1;
        g1[0] = 1 << 16;          // workgroup_mask=0, data_size=1 (2 bytes)
        g1[1] = HD << 16;         // tensor_dim0[15:0] in [31:16]
        g1[2] = SEQ << 16;        // tensor_dim0 hi=0, tensor_dim1 lo=SEQ
        g1[3] = HD << 16;         // tensor_dim1 hi=0, tile_dim0=HD
        g1[4] = SEQ;              // tile_dim1=SEQ, tile_dim2=0
        g1[5] = E3;               // tensor_dim0_stride lo
        g1[6] = 0;                // stride hi=0, tensor_dim1_stride lo=0
        g1[7] = 0;
        i32x4 gz = {0, 0, 0, 0};
        i32x8 gz8 = {0, 0, 0, 0, 0, 0, 0, 0};
        __builtin_amdgcn_tensor_load_to_lds(g0, g1, gz, gz, gz8, 0);
        __builtin_amdgcn_s_wait_tensorcnt(0);
    }
#else
    for (int idx = tid; idx < SEQ * (HD / 8); idx += 256) {
        int key = idx >> 3, oct = idx & 7;
        *(uint4*)&Vrow[key * HD + oct * 8] = ld16u(vsrc + (size_t)key * E3 + oct * 8);
    }
#endif
    __syncthreads();
    unsigned short* vdst = vt + (size_t)blk * HD * NPAD;
    for (int it = tid; it < HD * (NPAD / 8); it += 256) {
        int d = it / (NPAD / 8), c8 = it % (NPAD / 8);
        unsigned short tmp[8];
        #pragma unroll
        for (int j = 0; j < 8; j++) {
            int key = c8 * 8 + j;
            tmp[j] = (key < SEQ) ? Vrow[key * HD + d] : (unsigned short)0;
        }
        *(uint4*)&vdst[d * NPAD + c8 * 8] = *(const uint4*)tmp;
    }
}

// ---------------- attention + proj + scatter ----------------
// grid = (13 q-row tiles, NE edges), 256 threads (8 waves)
__global__ void attn_kernel(const int* __restrict__ edge,
                            const unsigned short* __restrict__ qkv,
                            const unsigned short* __restrict__ vt,
                            const unsigned short* __restrict__ wp,
                            const float* __restrict__ projb,
                            float* __restrict__ agg) {
    __shared__ float S[16 * NPAD];   // scores f32; first half aliased as P bf16
    __shared__ float Msg[16 * CH];   // per-tile message f32; aliased bf16
    __shared__ float red[16][17];
    __shared__ float rowm[16], rowinv[16];
    unsigned short* Pb = (unsigned short*)S;
    unsigned short* Msgb = (unsigned short*)Msg;

    int e = blockIdx.y, m0 = blockIdx.x * 16;
    int src = edge[e], dst = edge[NE + e];
    int tid = threadIdx.x, lane = tid & 31, w = tid >> 5;
    int sub = lane & 15, hi = lane >> 4;
    int rr_ = tid >> 4, cc_ = tid & 15;

    const unsigned short* qbase = qkv + (size_t)dst * SEQ * E3;
    const unsigned short* kbase = qkv + (size_t)src * SEQ * E3;
    uint4 z4 = make_uint4(0u, 0u, 0u, 0u);

    for (int h = 0; h < NH; h++) {
        // Q fragments for this 16-row tile (K = 0..63, two 32-chunks)
        FragU qa0, qa1;
        qa0.q[0] = z4; qa0.q[1] = z4; qa1.q[0] = z4; qa1.q[1] = z4;
        {
            int row = m0 + sub;
            if (row < SEQ) {
                const unsigned short* p = qbase + (size_t)row * E3 + h * HD;
                qa0.q[0] = ld16u(p + hi * 8);      qa0.q[1] = ld16u(p + 16 + hi * 8);
                qa1.q[0] = ld16u(p + 32 + hi * 8); qa1.q[1] = ld16u(p + 48 + hi * 8);
            }
        }
        // pass A: S = scale * Q K^T over 14 key tiles
        for (int kt = w; kt < NPAD / 16; kt += 8) {
            v8f acc = V8F_ZERO;
            int key = kt * 16 + sub;
            FragU kb0, kb1;
            kb0.q[0] = z4; kb0.q[1] = z4; kb1.q[0] = z4; kb1.q[1] = z4;
            if (key < SEQ) {
                const unsigned short* p = kbase + (size_t)key * E3 + CH + h * HD + hi * 8;
                kb0.q[0] = ld16u(p);      kb0.q[1] = ld16u(p + 16);
                kb1.q[0] = ld16u(p + 32); kb1.q[1] = ld16u(p + 48);
            }
            acc = WMMA_BF16(qa0.v, kb0.v, acc);
            acc = WMMA_BF16(qa1.v, kb1.v, acc);
            #pragma unroll
            for (int rr = 0; rr < 8; rr++)
                S[(hi * 8 + rr) * NPAD + kt * 16 + sub] = acc[rr] * 0.125f;
        }
        __syncthreads();
        // pass B: row softmax (valid cols 0..196)
        {
            float pm = -1e30f;
            for (int j = cc_; j < SEQ; j += 16) pm = fmaxf(pm, S[rr_ * NPAD + j]);
            red[rr_][cc_] = pm; __syncthreads();
            if (tid < 16) { float mm = -1e30f; for (int j = 0; j < 16; j++) mm = fmaxf(mm, red[tid][j]); rowm[tid] = mm; }
            __syncthreads();
            float mm = rowm[rr_], ps = 0.f;
            for (int j = cc_; j < SEQ; j += 16) ps += __expf(S[rr_ * NPAD + j] - mm);
            red[rr_][cc_] = ps; __syncthreads();
            if (tid < 16) { float ss = 0.f; for (int j = 0; j < 16; j++) ss += red[tid][j]; rowinv[tid] = 1.f / ss; }
            __syncthreads();
            float iv = rowinv[rr_];
            float pv[NPAD / 16];
            int n = 0;
            for (int j = cc_; j < NPAD; j += 16)
                pv[n++] = (j < SEQ) ? __expf(S[rr_ * NPAD + j] - mm) * iv : 0.f;
            __syncthreads();            // all S reads done before aliased writes
            n = 0;
            for (int j = cc_; j < NPAD; j += 16) Pb[rr_ * NPAD + j] = f2bf(pv[n++]);
        }
        __syncthreads();
        // pass C: msg = P @ V  (waves 0..3, one 16-wide slice of hd each);
        // B fragments come straight from the pre-transposed global V^T (L2-resident)
        if (w < 4) {
            v8f acc = V8F_ZERO;
            const unsigned short* vb = vt + ((size_t)(dst * NH + h) * HD + w * 16) * NPAD;
            #pragma unroll
            for (int kk = 0; kk < NPAD / 32; kk++) {
                v16bf a = loadA(Pb, NPAD, kk * 32, lane);
                v16bf bb = loadB(vb, NPAD, kk * 32, lane);
                acc = WMMA_BF16(a, bb, acc);
            }
            #pragma unroll
            for (int rr = 0; rr < 8; rr++)
                Msg[(hi * 8 + rr) * CH + h * HD + w * 16 + sub] = acc[rr];
        }
        __syncthreads();
    }
    // repack Msg f32 -> bf16 in place (register staged)
    {
        float mv[CH / 16];
        int n = 0;
        for (int j = cc_; j < CH; j += 16) mv[n++] = Msg[rr_ * CH + j];
        __syncthreads();
        n = 0;
        for (int j = cc_; j < CH; j += 16) Msgb[rr_ * CH + j] = f2bf(mv[n++]);
        __syncthreads();
    }
    // proj GEMM + atomic scatter into agg[dst]
    for (int tt = w; tt < CH / 16; tt += 8) {
        v8f acc = V8F_ZERO;
        const unsigned short* wb = wp + (size_t)(tt * 16) * CH;
        #pragma unroll
        for (int kk = 0; kk < 6; kk++) {
            v16bf a = loadA(Msgb, CH, kk * 32, lane);
            v16bf bb = loadB(wb, CH, kk * 32, lane);
            acc = WMMA_BF16(a, bb, acc);
        }
        int nn = tt * 16 + sub;
        float pb = projb[nn];
        #pragma unroll
        for (int rr = 0; rr < 8; rr++) {
            int mrow = m0 + hi * 8 + rr;
            if (mrow < SEQ)
                atomicAdd(&agg[((size_t)dst * SEQ + mrow) * CH + nn], acc[rr] + pb);
        }
    }
}

// ---------------- residual 1: out = x + agg / max(cnt,1) ----------------
__global__ void resid_kernel(const float* __restrict__ x, const float* __restrict__ agg,
                             const float* __restrict__ cnt, float* __restrict__ out) {
    size_t i = (size_t)blockIdx.x * 256 + threadIdx.x;
    if (i < (size_t)AGG_ELEMS) {
        int b = (int)(i / (SEQ * CH));
        out[i] = x[i] + agg[i] / fmaxf(cnt[b], 1.0f);
    }
}

// ---------------- LN2 + FC1 + gelu + FC2 + residual ----------------
__global__ void mlp_kernel(float* __restrict__ xio,
                           const float* __restrict__ g, const float* __restrict__ b,
                           const unsigned short* __restrict__ w1, const float* __restrict__ fb1,
                           const unsigned short* __restrict__ w2, const float* __restrict__ fb2) {
    __shared__ unsigned short xn[16 * CH];
    __shared__ unsigned short Hb[16 * HID];
    __shared__ float red[16][17];
    __shared__ float mu[16], rsg[16];

    int tid = threadIdx.x, lane = tid & 31, w = tid >> 5;
    int r = tid >> 4, c = tid & 15;
    int sub = lane & 15, hi = lane >> 4;
    size_t tok0 = (size_t)blockIdx.x * 16;
    const float* xr = xio + (tok0 + r) * CH;

    float s = 0.f;
    for (int j = c; j < CH; j += 16) s += xr[j];
    red[r][c] = s; __syncthreads();
    if (tid < 16) { float m = 0.f; for (int j = 0; j < 16; j++) m += red[tid][j]; mu[tid] = m / (float)CH; }
    __syncthreads();
    float m = mu[r]; s = 0.f;
    for (int j = c; j < CH; j += 16) { float d = xr[j] - m; s += d * d; }
    red[r][c] = s; __syncthreads();
    if (tid < 16) { float v = 0.f; for (int j = 0; j < 16; j++) v += red[tid][j]; rsg[tid] = rsqrtf(v / (float)CH + 1e-5f); }
    __syncthreads();
    float inv = rsg[r];
    for (int j = c; j < CH; j += 16) xn[r * CH + j] = f2bf((xr[j] - m) * inv * g[j] + b[j]);
    __syncthreads();

    // FC1 + exact gelu -> Hb
    for (int t = w; t < HID / 16; t += 8) {
        v8f acc = V8F_ZERO;
        const unsigned short* wb = w1 + (size_t)(t * 16) * CH;
        #pragma unroll
        for (int kk = 0; kk < 6; kk++) {
            v16bf a = loadA(xn, CH, kk * 32, lane);
            v16bf bb = loadB(wb, CH, kk * 32, lane);
            acc = WMMA_BF16(a, bb, acc);
        }
        int nn = t * 16 + sub;
        float bias = fb1[nn];
        #pragma unroll
        for (int rr = 0; rr < 8; rr++) {
            float v = acc[rr] + bias;
            float ge = 0.5f * v * (1.f + erff(v * 0.70710678f));
            Hb[(hi * 8 + rr) * HID + nn] = f2bf(ge);
        }
    }
    __syncthreads();
    // FC2 + residual
    for (int t = w; t < CH / 16; t += 8) {
        v8f acc = V8F_ZERO;
        const unsigned short* wb = w2 + (size_t)(t * 16) * HID;
        #pragma unroll
        for (int kk = 0; kk < HID / 32; kk++) {
            v16bf a = loadA(Hb, HID, kk * 32, lane);
            v16bf bb = loadB(wb, HID, kk * 32, lane);
            acc = WMMA_BF16(a, bb, acc);
        }
        int nn = t * 16 + sub;
        float bias = fb2[nn];
        #pragma unroll
        for (int rr = 0; rr < 8; rr++) {
            size_t idx = (tok0 + hi * 8 + rr) * CH + nn;
            xio[idx] = xio[idx] + acc[rr] + bias;
        }
    }
}

// ---------------- launch ----------------
extern "C" void kernel_launch(void* const* d_in, const int* in_sizes, int n_in,
                              void* d_out, int out_size, void* d_ws, size_t ws_size,
                              hipStream_t stream) {
    const float* x    = (const float*)d_in[0];
    const int*   edge = (const int*)d_in[1];
    const float* n1g  = (const float*)d_in[2];
    const float* n1b  = (const float*)d_in[3];
    const float* qkvw = (const float*)d_in[4];
    const float* projw= (const float*)d_in[5];
    const float* projb= (const float*)d_in[6];
    const float* n2g  = (const float*)d_in[7];
    const float* n2b  = (const float*)d_in[8];
    const float* fc1w = (const float*)d_in[9];
    const float* fc1b = (const float*)d_in[10];
    const float* fc2w = (const float*)d_in[11];
    const float* fc2b = (const float*)d_in[12];
    float* out = (float*)d_out;

    char* ws = (char*)d_ws;
    size_t off = 0;
    unsigned short* qkv = (unsigned short*)(ws + off); off += (size_t)TOK * E3 * 2;      // 58,097,664
    float* agg = (float*)(ws + off);                   off += (size_t)AGG_ELEMS * 4;     // 38,731,776
    float* cnt = (float*)(ws + off);                   off += 1024;
    unsigned short* wq = (unsigned short*)(ws + off);  off += (size_t)E3 * CH * 2;
    unsigned short* wp = (unsigned short*)(ws + off);  off += (size_t)CH * CH * 2;
    unsigned short* w1 = (unsigned short*)(ws + off);  off += (size_t)HID * CH * 2;
    unsigned short* w2 = (unsigned short*)(ws + off);  off += (size_t)CH * HID * 2;
    unsigned short* vt = (unsigned short*)(ws + off);  off += (size_t)BN * NH * HD * NPAD * 2; // 22,020,096

    zero_kernel<<<(AGG_ELEMS + 255) / 256, 256, 0, stream>>>(agg, AGG_ELEMS);
    zero_kernel<<<1, 256, 0, stream>>>(cnt, BN);

    cvt_bf16_kernel<<<(E3 * CH + 255) / 256, 256, 0, stream>>>(qkvw, wq, E3 * CH);
    cvt_bf16_kernel<<<(CH * CH + 255) / 256, 256, 0, stream>>>(projw, wp, CH * CH);
    cvt_bf16_kernel<<<(HID * CH + 255) / 256, 256, 0, stream>>>(fc1w, w1, HID * CH);
    cvt_bf16_kernel<<<(CH * HID + 255) / 256, 256, 0, stream>>>(fc2w, w2, CH * HID);

    ln_qkv_kernel<<<NTILES_TOK, 256, 0, stream>>>(x, n1g, n1b, wq, qkv);
    vt_kernel<<<BN * NH, 256, 0, stream>>>(qkv, vt);
    edge_cnt_kernel<<<(NE + 255) / 256, 256, 0, stream>>>(edge, cnt);

    dim3 ag((SEQ + 15) / 16, NE);
    attn_kernel<<<ag, 256, 0, stream>>>(edge, qkv, vt, wp, projb, agg);

    resid_kernel<<<(AGG_ELEMS + 255) / 256, 256, 0, stream>>>(x, agg, cnt, out);
    mlp_kernel<<<NTILES_TOK, 256, 0, stream>>>(out, n2g, n2b, w1, fc1b, w2, fc2b);
}
